// AttentionLayer_43215960933013
// MI455X (gfx1250) — matrix operand order repaired and checked by reading, not compiled, not run on hardware
//
#include <hip/hip_runtime.h>
#include <math.h>

// ---------------------------------------------------------------------------
// Causal single-head attention (B=4, T=2048, C=H=768) for MI455X / gfx1250.
// bf16 WMMA (v_wmma_f32_16x16x32_bf16) with fp32 accumulation.
// LDS-staged B operands shared across the 4 waves of each block.
// ---------------------------------------------------------------------------

#define BATCH 4
#define SEQ   2048
#define CDIM  768
#define HDIM  768
#define MTOT  (BATCH * SEQ)          // 8192 rows
#define SCALE 0.036084391824351615f  // 1/sqrt(768)

typedef __attribute__((ext_vector_type(16))) __bf16 v16bf;
typedef __attribute__((ext_vector_type(8)))  float  v8f;

union BFrag {
    v16bf v;
    uint4 q[2];
    unsigned short u[16];
};

__device__ __forceinline__ unsigned short f2bf(float f) {
    unsigned int u = __builtin_bit_cast(unsigned int, f);
    unsigned int r = u + 0x7FFFu + ((u >> 16) & 1u);   // round-to-nearest-even
    return (unsigned short)(r >> 16);
}

__device__ __forceinline__ v8f vzero8() {
    v8f z;
    for (int i = 0; i < 8; ++i) z[i] = 0.0f;
    return z;
}

// A fragment (16x32 bf16, M x K): lane (l&15) = row m; half-wave selects K runs
// {half*8 .. +7} in elems 0..7 and {16+half*8 .. +7} in elems 8..15.
__device__ __forceinline__ v16bf load_a_frag(const unsigned short* mat, int stride,
                                             int m0, int k0) {
    int lane = threadIdx.x & 31;
    const unsigned short* p =
        mat + (size_t)(m0 + (lane & 15)) * stride + k0 + ((lane >> 4) << 3);
    BFrag f;
    f.q[0] = *(const uint4*)p;          // K = k0+half*8    .. +7
    f.q[1] = *(const uint4*)(p + 16);   // K = k0+16+half*8 .. +7
    return f.v;
}

// B fragment (32x16 bf16, K x N) from a [N][K]-major (transposed) matrix:
// lane (l&15) = column n; elems 0..7 = K k0+half*8.., elems 8..15 = K k0+16+half*8..
__device__ __forceinline__ v16bf load_b_frag(const unsigned short* mat, int stride,
                                             int n0, int k0) {
    int lane = threadIdx.x & 31;
    const unsigned short* p =
        mat + (size_t)(n0 + (lane & 15)) * stride + k0 + ((lane >> 4) << 3);
    BFrag f;
    f.q[0] = *(const uint4*)p;
    f.q[1] = *(const uint4*)(p + 16);
    return f.v;
}

// Block-cooperative (128 threads) global->LDS tile copy, uint4 granularity.
// cols must be a multiple of 8 elements; all bases 16B aligned.
__device__ __forceinline__ void copy_tile_g2l(unsigned short* __restrict__ dst,
                                              const unsigned short* __restrict__ src,
                                              int rows, int cols, int src_stride) {
    int c4 = cols >> 3;
    int total = rows * c4;
    for (int idx = threadIdx.x; idx < total; idx += 128) {
        int r = idx / c4;
        int c = idx - r * c4;
        *(uint4*)(dst + (size_t)r * cols + c * 8) =
            *(const uint4*)(src + (size_t)r * src_stride + c * 8);
    }
}

#define WMMA_BF16(a, b, c) \
    __builtin_amdgcn_wmma_f32_16x16x32_bf16(false, (a), false, (b), (short)0, (c), false, false)

// --------------------------- prep kernels ----------------------------------

__global__ void k_cvt_x(const float* __restrict__ x, unsigned short* __restrict__ xb, int n) {
    int i = blockIdx.x * 256 + threadIdx.x;
    if (i < n) xb[i] = f2bf(x[i]);
}

// Wt[z][n][k] = W_z[k][n] in bf16 (B-operand friendly layout)
__global__ void k_cvt_w(const float* __restrict__ Wq, const float* __restrict__ Wk,
                        const float* __restrict__ Wv, unsigned short* __restrict__ wt) {
    int z = blockIdx.z;
    const float* W = (z == 0) ? Wq : (z == 1) ? Wk : Wv;
    unsigned short* o = wt + (size_t)z * CDIM * HDIM;
    int i = blockIdx.x * 256 + threadIdx.x;
    if (i < CDIM * HDIM) {
        int n = i / CDIM, k = i - n * CDIM;
        o[(size_t)n * CDIM + k] = f2bf(W[(size_t)k * HDIM + n]);
    }
}

// --------------------------- fused QKV GEMM --------------------------------
// grid (MTOT/128, HDIM/64, 3); block 128 = 4 waves; wave w owns 32 rows.
// Weight tile staged in LDS (64 cols x 256 k, 32KB), shared by all waves.
// q is pre-scaled by 1/sqrt(H); v is stored transposed [B][H][T].
__global__ __launch_bounds__(128) void k_qkv(const unsigned short* __restrict__ xb,
                                             const unsigned short* __restrict__ wt,
                                             const float* __restrict__ bq,
                                             const float* __restrict__ bk,
                                             const float* __restrict__ bv,
                                             unsigned short* __restrict__ qb,
                                             unsigned short* __restrict__ kb,
                                             unsigned short* __restrict__ vt) {
    __shared__ __align__(16) unsigned short wstage[64 * 256];   // 32 KB

    int z    = blockIdx.z;
    int r    = blockIdx.x * 128 + (threadIdx.x >> 5) * 32;      // wave's first row
    int n0b  = blockIdx.y * 64;
    int lane = threadIdx.x & 31;
    int half = lane >> 4, nl = lane & 15;

    const unsigned short* w = wt + (size_t)z * CDIM * HDIM;
    v8f acc[8];
    for (int t = 0; t < 8; ++t) acc[t] = vzero8();

    for (int c0 = 0; c0 < CDIM; c0 += 256) {
        __syncthreads();
        copy_tile_g2l(wstage, w + (size_t)n0b * CDIM + c0, 64, 256, CDIM);
        __syncthreads();
        for (int k0 = 0; k0 < 256; k0 += 32) {
            v16bf a0 = load_a_frag(xb, CDIM, r,      c0 + k0);
            v16bf a1 = load_a_frag(xb, CDIM, r + 16, c0 + k0);
            for (int t = 0; t < 4; ++t) {
                v16bf b = load_b_frag(wstage, 256, t * 16, k0);
                acc[t]     = WMMA_BF16(a0, b, acc[t]);
                acc[4 + t] = WMMA_BF16(a1, b, acc[4 + t]);
            }
        }
    }

    const float* bias = (z == 0) ? bq : (z == 1) ? bk : bv;
    for (int st = 0; st < 2; ++st) {
        int rbase = r + st * 16;
        for (int t = 0; t < 4; ++t) {
            int n = n0b + t * 16 + nl;
            float bn = bias[n];
            for (int j = 0; j < 8; ++j) {
                int m = rbase + half * 8 + j;   // global token row
                float v = acc[st * 4 + t][j] + bn;
                if (z == 0) {
                    qb[(size_t)m * HDIM + n] = f2bf(v * SCALE);
                } else if (z == 1) {
                    kb[(size_t)m * HDIM + n] = f2bf(v);
                } else {
                    int b_ = m >> 11, tt = m & (SEQ - 1);
                    vt[((size_t)b_ * HDIM + n) * SEQ + tt] = f2bf(v);
                }
            }
        }
    }
}

// --------------------- softmax row statistics (m, l) -----------------------
// grid (MTOT/64); wave owns 16 query rows; K-block staged in LDS (shared).
__global__ __launch_bounds__(128) void k_stats(const unsigned short* __restrict__ qb,
                                               const unsigned short* __restrict__ kb,
                                               float* __restrict__ mbuf,
                                               float* __restrict__ lbuf) {
    __shared__ __align__(16) unsigned short kstage[64 * 256];   // 32 KB

    int qblk  = blockIdx.x;
    int batch = (qblk * 64) >> 11;
    int t0    = (qblk * 64) & (SEQ - 1);
    int wave  = threadIdx.x >> 5;
    int lane  = threadIdx.x & 31;
    int half  = lane >> 4, nl = lane & 15;
    int r0    = qblk * 64 + wave * 16;

    const unsigned short* kbase = kb + (size_t)batch * SEQ * HDIM;
    const float NEGINF = -__builtin_inff();
    float m[8], l[8];
    for (int j = 0; j < 8; ++j) { m[j] = NEGINF; l[j] = 0.0f; }

    int jmax = t0 / 64;   // causal bound (same for all waves in the block)
    for (int kblk = 0; kblk <= jmax; ++kblk) {
        int s0 = kblk * 64;
        v8f acc[4];
        for (int t = 0; t < 4; ++t) acc[t] = vzero8();
        for (int c0 = 0; c0 < HDIM; c0 += 256) {
            __syncthreads();
            copy_tile_g2l(kstage, kbase + (size_t)s0 * HDIM + c0, 64, 256, HDIM);
            __syncthreads();
            for (int k0 = 0; k0 < 256; k0 += 32) {
                v16bf a = load_a_frag(qb, HDIM, r0, c0 + k0);
                for (int t = 0; t < 4; ++t) {
                    v16bf b = load_b_frag(kstage, 256, t * 16, k0);
                    acc[t] = WMMA_BF16(a, b, acc[t]);
                }
            }
        }
        for (int j = 0; j < 8; ++j) {
            int trow = t0 + wave * 16 + half * 8 + j;   // local query pos
            float mb = NEGINF;
            for (int t = 0; t < 4; ++t) {
                int s = s0 + t * 16 + nl;
                float v = acc[t][j];
                if (s > trow || v == 0.0f) v = NEGINF;  // tril + zero quirk
                mb = fmaxf(mb, v);
            }
            for (int off = 1; off < 16; off <<= 1)
                mb = fmaxf(mb, __shfl_xor(mb, off, 32));
            float mn = fmaxf(m[j], mb);
            float sum = 0.0f;
            for (int t = 0; t < 4; ++t) {
                int s = s0 + t * 16 + nl;
                float v = acc[t][j];
                if (!(s > trow || v == 0.0f)) sum += __expf(v - mn);
            }
            for (int off = 1; off < 16; off <<= 1)
                sum += __shfl_xor(sum, off, 32);
            float so = (m[j] == NEGINF) ? 0.0f : __expf(m[j] - mn);
            l[j] = l[j] * so + sum;
            m[j] = mn;
        }
    }
    if (nl == 0) {
        for (int j = 0; j < 8; ++j) {
            int row = r0 + half * 8 + j;
            mbuf[row] = m[j];
            lbuf[row] = l[j];
        }
    }
}

// --------------------------- O = P @ V -------------------------------------
// grid (MTOT/64, HDIM/128); recompute S via WMMA (K staged in LDS),
// P = exp(S-m)/l, P round-trips through per-wave LDS for layout conversion,
// then P@V with V block (128x64) staged in LDS.  Static LDS total: 56 KB.
__global__ __launch_bounds__(128) void k_out(const unsigned short* __restrict__ qb,
                                             const unsigned short* __restrict__ kb,
                                             const unsigned short* __restrict__ vt,
                                             const float* __restrict__ mbuf,
                                             const float* __restrict__ lbuf,
                                             float* __restrict__ out) {
    __shared__ __align__(16) unsigned short kstage[64 * 256];   // 32 KB
    __shared__ __align__(16) unsigned short vstage[128 * 64];   // 16 KB
    __shared__ __align__(16) unsigned short plds[4][16 * 64];   //  8 KB

    int qblk  = blockIdx.x;
    int d0    = blockIdx.y * 128;
    int batch = (qblk * 64) >> 11;
    int t0    = (qblk * 64) & (SEQ - 1);
    int wave  = threadIdx.x >> 5;
    int lane  = threadIdx.x & 31;
    int half  = lane >> 4, nl = lane & 15;
    int r0    = qblk * 64 + wave * 16;

    const unsigned short* kbase = kb + (size_t)batch * SEQ * HDIM;
    const unsigned short* vbase = vt + (size_t)batch * HDIM * SEQ;
    const float NEGINF = -__builtin_inff();

    float mrow[8], linv[8];
    for (int j = 0; j < 8; ++j) {
        int row = r0 + half * 8 + j;
        mrow[j] = mbuf[row];
        float L = lbuf[row];
        linv[j] = (L > 0.0f) ? 1.0f / L : 0.0f;
    }

    v8f o[8];
    for (int t = 0; t < 8; ++t) o[t] = vzero8();

    int jmax = t0 / 64;
    for (int kblk = 0; kblk <= jmax; ++kblk) {
        int s0 = kblk * 64;

        // ---- S = q k^T for this 64-key block (K staged in LDS) ----
        v8f acc[4];
        for (int t = 0; t < 4; ++t) acc[t] = vzero8();
        for (int c0 = 0; c0 < HDIM; c0 += 256) {
            __syncthreads();
            copy_tile_g2l(kstage, kbase + (size_t)s0 * HDIM + c0, 64, 256, HDIM);
            __syncthreads();
            for (int k0 = 0; k0 < 256; k0 += 32) {
                v16bf a = load_a_frag(qb, HDIM, r0, c0 + k0);
                for (int t = 0; t < 4; ++t) {
                    v16bf b = load_b_frag(kstage, 256, t * 16, k0);
                    acc[t] = WMMA_BF16(a, b, acc[t]);
                }
            }
        }

        // ---- P tile to per-wave LDS (same-wave only: DS in-order) ----
        for (int t = 0; t < 4; ++t) {
            for (int j = 0; j < 8; ++j) {
                int trow = t0 + wave * 16 + half * 8 + j;
                int s = s0 + t * 16 + nl;
                float v = acc[t][j];
                float p = (s > trow || v == 0.0f) ? 0.0f
                                                  : __expf(v - mrow[j]) * linv[j];
                plds[wave][(half * 8 + j) * 64 + t * 16 + nl] = f2bf(p);
            }
        }

        // ---- V block (128 d x 64 s) to LDS, shared by all waves ----
        __syncthreads();
        copy_tile_g2l(vstage, vbase + (size_t)d0 * SEQ + s0, 128, 64, SEQ);
        __syncthreads();

        // ---- O += P @ V ----
        for (int sc = 0; sc < 2; ++sc) {
            v16bf pa = load_a_frag(&plds[wave][0], 64, 0, sc * 32);
            for (int dt = 0; dt < 8; ++dt) {
                v16bf b = load_b_frag(vstage, 64, dt * 16, sc * 32);
                o[dt] = WMMA_BF16(pa, b, o[dt]);
            }
        }
    }

    for (int dt = 0; dt < 8; ++dt) {
        for (int j = 0; j < 8; ++j) {
            int row = r0 + half * 8 + j;
            out[(size_t)row * HDIM + d0 + dt * 16 + nl] = o[dt][j];
        }
    }
}

// --------------------------- launcher --------------------------------------

extern "C" void kernel_launch(void* const* d_in, const int* in_sizes, int n_in,
                              void* d_out, int out_size, void* d_ws, size_t ws_size,
                              hipStream_t stream) {
    (void)in_sizes; (void)n_in; (void)out_size; (void)ws_size;
    const float* x  = (const float*)d_in[0];
    const float* Wq = (const float*)d_in[1];
    const float* bq = (const float*)d_in[2];
    const float* Wk = (const float*)d_in[3];
    const float* bk = (const float*)d_in[4];
    const float* Wv = (const float*)d_in[5];
    const float* bv = (const float*)d_in[6];

    // workspace layout (bf16 elements unless noted); every region 16B-aligned
    unsigned short* xb   = (unsigned short*)d_ws;                    // [8192][768]
    unsigned short* wt   = xb   + (size_t)MTOT * CDIM;               // 3x[768][768] (W^T)
    unsigned short* qbuf = wt   + (size_t)3 * CDIM * HDIM;           // [8192][768] (scaled)
    unsigned short* kbuf = qbuf + (size_t)MTOT * HDIM;               // [8192][768]
    unsigned short* vtb  = kbuf + (size_t)MTOT * HDIM;               // [4][768][2048] (V^T)
    float* mbuf = (float*)(vtb + (size_t)MTOT * HDIM);               // [8192]
    float* lbuf = mbuf + MTOT;                                       // [8192]

    k_cvt_x<<<(MTOT * CDIM + 255) / 256, 256, 0, stream>>>(x, xb, MTOT * CDIM);
    k_cvt_w<<<dim3((CDIM * HDIM + 255) / 256, 1, 3), 256, 0, stream>>>(Wq, Wk, Wv, wt);
    k_qkv<<<dim3(MTOT / 128, HDIM / 64, 3), 128, 0, stream>>>(xb, wt, bq, bk, bv,
                                                              qbuf, kbuf, vtb);
    k_stats<<<MTOT / 64, 128, 0, stream>>>(qbuf, kbuf, mbuf, lbuf);
    k_out<<<dim3(MTOT / 64, HDIM / 128), 128, 0, stream>>>(qbuf, kbuf, vtb,
                                                           mbuf, lbuf, (float*)d_out);
}
